// TernaryLinear_89936615178659
// MI455X (gfx1250) — 8-bit matrix operand order repaired and checked by reading, compiled, not run on hardware
//
#include <hip/hip_runtime.h>

typedef __attribute__((ext_vector_type(16))) int   v16i;
typedef __attribute__((ext_vector_type(8)))  float v8f;

static constexpr int   kN     = 8192;
static constexpr int   kIn    = 4096;
static constexpr int   kOut   = 4096;
static constexpr float kRatio = 0.05f;
static constexpr int   kParts = 1024;

__device__ float g_partials[kParts];
__device__ float g_scale;

__global__ void k_absmean_partial(const float* __restrict__ w) {
    __shared__ float sm[256];
    const int tid = threadIdx.x;
    const size_t n4     = (size_t)kIn * kOut / 4;
    const size_t stride = (size_t)gridDim.x * 256;
    const float4* w4 = (const float4*)w;
    float s = 0.f;
    for (size_t i = (size_t)blockIdx.x * 256 + tid; i < n4; i += stride) {
        float4 v = w4[i];
        s += fabsf(v.x) + fabsf(v.y) + fabsf(v.z) + fabsf(v.w);
    }
    sm[tid] = s;
    __syncthreads();
    for (int off = 128; off > 0; off >>= 1) {
        if (tid < off) sm[tid] += sm[tid + off];
        __syncthreads();
    }
    if (tid == 0) g_partials[blockIdx.x] = sm[0];
}

__global__ void k_absmean_final() {
    __shared__ float sm[256];
    const int tid = threadIdx.x;
    float s = 0.f;
    for (int i = tid; i < kParts; i += 256) s += g_partials[i];
    sm[tid] = s;
    __syncthreads();
    for (int off = 128; off > 0; off >>= 1) {
        if (tid < off) sm[tid] += sm[tid + off];
        __syncthreads();
    }
    if (tid == 0) g_scale = sm[0] * (1.0f / ((float)kIn * (float)kOut));
}

__global__ void k_ternarize_fp8(const float* __restrict__ w, unsigned int* __restrict__ wp) {
    const float thr = g_scale * kRatio;
    const size_t i = (size_t)blockIdx.x * blockDim.x + threadIdx.x;
    const float4 v = ((const float4*)w)[i];
    unsigned int b0 = (v.x > thr) ? 0x38u : ((v.x < -thr) ? 0xB8u : 0x00u);
    unsigned int b1 = (v.y > thr) ? 0x38u : ((v.y < -thr) ? 0xB8u : 0x00u);
    unsigned int b2 = (v.z > thr) ? 0x38u : ((v.z < -thr) ? 0xB8u : 0x00u);
    unsigned int b3 = (v.w > thr) ? 0x38u : ((v.w < -thr) ? 0xB8u : 0x00u);
    wp[i] = b0 | (b1 << 8) | (b2 << 16) | (b3 << 24);
}

__global__ void k_digitize(const float* __restrict__ x,
                           unsigned int* __restrict__ q0,
                           unsigned int* __restrict__ q1,
                           unsigned int* __restrict__ q2) {
    const size_t i = (size_t)blockIdx.x * blockDim.x + threadIdx.x;
    const float4 f = ((const float4*)x)[i];
    int d0 = 0, d1 = 0, d2 = 0;
    d0 = __builtin_amdgcn_cvt_pk_fp8_f32(f.x, f.y, d0, false);
    d0 = __builtin_amdgcn_cvt_pk_fp8_f32(f.z, f.w, d0, true);
    float r0 = f.x - __builtin_amdgcn_cvt_f32_fp8(d0, 0);
    float r1 = f.y - __builtin_amdgcn_cvt_f32_fp8(d0, 1);
    float r2 = f.z - __builtin_amdgcn_cvt_f32_fp8(d0, 2);
    float r3 = f.w - __builtin_amdgcn_cvt_f32_fp8(d0, 3);
    d1 = __builtin_amdgcn_cvt_pk_fp8_f32(r0, r1, d1, false);
    d1 = __builtin_amdgcn_cvt_pk_fp8_f32(r2, r3, d1, true);
    float s0 = r0 - __builtin_amdgcn_cvt_f32_fp8(d1, 0);
    float s1 = r1 - __builtin_amdgcn_cvt_f32_fp8(d1, 1);
    float s2 = r2 - __builtin_amdgcn_cvt_f32_fp8(d1, 2);
    float s3 = r3 - __builtin_amdgcn_cvt_f32_fp8(d1, 3);
    d2 = __builtin_amdgcn_cvt_pk_bf8_f32(s0, s1, d2, false);
    d2 = __builtin_amdgcn_cvt_pk_bf8_f32(s2, s3, d2, true);
    q0[i] = (unsigned int)d0;
    q1[i] = (unsigned int)d1;
    q2[i] = (unsigned int)d2;
}

__device__ __forceinline__ v16i loadA(const unsigned char* rowbase, int k0, int lhalf) {
    v16i a;
#pragma unroll
    for (int g = 0; g < 4; ++g) {
        int2 lo = *(const int2*)(rowbase + k0 +      16 * lhalf + 32 * (g >> 1) + 8 * (g & 1));
        int2 hi = *(const int2*)(rowbase + k0 + 64 + 16 * lhalf + 32 * (g >> 1) + 8 * (g & 1));
        a[2 * g]     = lo.x;  a[2 * g + 1] = lo.y;
        a[8 + 2 * g] = hi.x;  a[9 + 2 * g] = hi.y;
    }
    return a;
}

__device__ __forceinline__ v16i loadB(const unsigned char* colbase, int k0, int lhalf) {
    v16i b;
#pragma unroll
    for (int q = 0; q < 4; ++q) {
        int4 t = *(const int4*)(colbase + k0 + 32 * q + 16 * lhalf);
        b[4 * q] = t.x; b[4 * q + 1] = t.y; b[4 * q + 2] = t.z; b[4 * q + 3] = t.w;
    }
    return b;
}

__launch_bounds__(256, 1)
__global__ void k_gemm_fp8(const unsigned char* __restrict__ p0,
                           const unsigned char* __restrict__ p1,
                           const unsigned char* __restrict__ p2,
                           const unsigned char* __restrict__ wp,
                           const float* __restrict__ bias,
                           float* __restrict__ out) {
    const int lane  = threadIdx.x & 31;
    const int wave  = threadIdx.x >> 5;
    const int lrow  = lane & 15;
    const int lhalf = lane >> 4;
    const int mBase = blockIdx.y * 128 + (wave >> 1) * 32;
    const int nBase = blockIdx.x * 128 + (wave & 1) * 64;

    v8f acc[2][4];
#pragma unroll
    for (int i = 0; i < 2; ++i)
#pragma unroll
        for (int j = 0; j < 4; ++j) { v8f z = {}; acc[i][j] = z; }

    const size_t rA0 = (size_t)(mBase + lrow) * kIn;
    const size_t rA1 = rA0 + (size_t)16 * kIn;
    const unsigned char* wc[4];
#pragma unroll
    for (int tn = 0; tn < 4; ++tn)
        wc[tn] = wp + (size_t)(nBase + tn * 16 + lrow) * kIn;

    for (int k0 = 0; k0 < kIn; k0 += 128) {
        v16i bm[4];
#pragma unroll
        for (int tn = 0; tn < 4; ++tn) bm[tn] = loadB(wc[tn], k0, lhalf);

        {
            v16i a0 = loadA(p0 + rA0, k0, lhalf);
            v16i a1 = loadA(p0 + rA1, k0, lhalf);
#pragma unroll
            for (int tn = 0; tn < 4; ++tn) {
                acc[0][tn] = __builtin_amdgcn_wmma_f32_16x16x128_fp8_fp8(
                    a0, bm[tn], (short)0, acc[0][tn], false, false);
                acc[1][tn] = __builtin_amdgcn_wmma_f32_16x16x128_fp8_fp8(
                    a1, bm[tn], (short)0, acc[1][tn], false, false);
            }
        }
        {
            v16i a0 = loadA(p1 + rA0, k0, lhalf);
            v16i a1 = loadA(p1 + rA1, k0, lhalf);
#pragma unroll
            for (int tn = 0; tn < 4; ++tn) {
                acc[0][tn] = __builtin_amdgcn_wmma_f32_16x16x128_fp8_fp8(
                    a0, bm[tn], (short)0, acc[0][tn], false, false);
                acc[1][tn] = __builtin_amdgcn_wmma_f32_16x16x128_fp8_fp8(
                    a1, bm[tn], (short)0, acc[1][tn], false, false);
            }
        }
        {
            v16i a0 = loadA(p2 + rA0, k0, lhalf);
            v16i a1 = loadA(p2 + rA1, k0, lhalf);
#pragma unroll
            for (int tn = 0; tn < 4; ++tn) {
                acc[0][tn] = __builtin_amdgcn_wmma_f32_16x16x128_bf8_fp8(
                    a0, bm[tn], (short)0, acc[0][tn], false, false);
                acc[1][tn] = __builtin_amdgcn_wmma_f32_16x16x128_bf8_fp8(
                    a1, bm[tn], (short)0, acc[1][tn], false, false);
            }
        }
    }

    const float s = g_scale;
#pragma unroll
    for (int tm = 0; tm < 2; ++tm) {
        const int row0 = mBase + tm * 16 + 8 * lhalf;
#pragma unroll
        for (int tn = 0; tn < 4; ++tn) {
            const int col = nBase + tn * 16 + lrow;
            const float bv = bias[col];
            float* op = out + (size_t)row0 * kOut + col;
#pragma unroll
            for (int v = 0; v < 8; ++v)
                op[(size_t)v * kOut] = acc[tm][tn][v] * s + bv;
        }
    }
}

__global__ void k_naive(const float* __restrict__ x, const float* __restrict__ w,
                        const float* __restrict__ bias, float* __restrict__ out) {
    const size_t idx = (size_t)blockIdx.x * blockDim.x + threadIdx.x;
    const int row = (int)(idx / kOut);
    const int col = (int)(idx % kOut);
    const float sc = g_scale, thr = sc * kRatio;
    const float* xp  = x + (size_t)row * kIn;
    const float* wpr = w + (size_t)col * kIn;
    float acc = 0.f;
    for (int k = 0; k < kIn; ++k) {
        float wv = wpr[k];
        float t = (wv > thr) ? 1.0f : ((wv < -thr) ? -1.0f : 0.0f);
        acc += xp[k] * t;
    }
    out[idx] = acc * sc + bias[col];
}

extern "C" void kernel_launch(void* const* d_in, const int* in_sizes, int n_in,
                              void* d_out, int out_size, void* d_ws, size_t ws_size,
                              hipStream_t stream) {
    (void)in_sizes; (void)n_in; (void)out_size;
    const float* x    = (const float*)d_in[0];
    const float* w    = (const float*)d_in[1];
    const float* bias = (const float*)d_in[2];
    float* out        = (float*)d_out;

    k_absmean_partial<<<kParts, 256, 0, stream>>>(w);
    k_absmean_final<<<1, 256, 0, stream>>>();

    const size_t planeBytes = (size_t)kN * kIn;
    const size_t wBytes     = (size_t)kOut * kIn;
    const size_t need       = 3 * planeBytes + wBytes;

    if (ws_size >= need) {
        unsigned char* base = (unsigned char*)d_ws;
        unsigned char* q0 = base;
        unsigned char* q1 = base + planeBytes;
        unsigned char* q2 = base + 2 * planeBytes;
        unsigned char* wq = base + 3 * planeBytes;

        const int ternBlocks = (int)(((size_t)kIn * kOut) / (4 * 256));
        k_ternarize_fp8<<<ternBlocks, 256, 0, stream>>>(w, (unsigned int*)wq);

        const int digBlocks = (int)(((size_t)kN * kIn) / (4 * 256));
        k_digitize<<<digBlocks, 256, 0, stream>>>(x, (unsigned int*)q0,
                                                  (unsigned int*)q1, (unsigned int*)q2);

        dim3 grid(kOut / 128, kN / 128);
        k_gemm_fp8<<<grid, 256, 0, stream>>>(q0, q1, q2, wq, bias, out);
    } else {
        k_naive<<<(int)(((size_t)kN * kOut) / 256), 256, 0, stream>>>(x, w, bias, out);
    }
}
